// MultiHeadAttention_39779987096081
// MI455X (gfx1250) — matrix operand order repaired and checked by reading, not compiled, not run on hardware
//
#include <hip/hip_runtime.h>

typedef __attribute__((ext_vector_type(16))) _Float16 v16h;
typedef __attribute__((ext_vector_type(8)))  float    v8f;
typedef __attribute__((ext_vector_type(4)))  float    v4f;

#define D_MODEL   1024
#define NUM_HEADS 16
#define DEPTH     64
#define BATCH     4
#define SEQ       2048
#define ROWS      (BATCH * SEQ)   /* 8192 */
#define LSTRIDE   2052            /* padded row stride (words); 2052*4 B is 16B-aligned */

// ---------------------------------------------------------------------------
// WMMA 16x16x32 f16 fragment loaders (wave32, contiguous k order: element s holds k = 16*(lane/16) + s)
// ---------------------------------------------------------------------------

// A-matrix 16x32 (MxK) from row-major f16 [M x ld], tile origin (m0, k0).
__device__ __forceinline__ v16h load_a_f16(const _Float16* __restrict__ src,
                                           int ld, int m0, int k0, int lane) {
  const int row = m0 + (lane & 15);
  const int kb  = (lane < 16) ? 0 : 16;
  const _Float16* p = src + (size_t)row * ld + k0 + kb;
  v16h a;
#pragma unroll
  for (int v = 0; v < 8; ++v) {
    const int ko = 2 * v;
    a[2 * v]     = p[ko];
    a[2 * v + 1] = p[ko + 1];
  }
  return a;
}

// Same A layout, converting from f32 source on the fly.
__device__ __forceinline__ v16h load_a_f32(const float* __restrict__ src,
                                           int ld, int m0, int k0, int lane) {
  const int row = m0 + (lane & 15);
  const int kb  = (lane < 16) ? 0 : 16;
  const float* p = src + (size_t)row * ld + k0 + kb;
  v16h a;
#pragma unroll
  for (int v = 0; v < 8; ++v) {
    const int ko = 2 * v;
    a[2 * v]     = (_Float16)p[ko];
    a[2 * v + 1] = (_Float16)p[ko + 1];
  }
  return a;
}

// B-matrix 32x16 (KxN) from row-major f16 [K x ld], tile origin (k0, n0).
__device__ __forceinline__ v16h load_b_f16(const _Float16* __restrict__ src,
                                           int ld, int k0, int n0, int lane) {
  const int col = n0 + (lane & 15);
  const int kb  = (lane < 16) ? 0 : 16;
  const _Float16* p = src + (size_t)(k0 + kb) * ld + col;
  v16h b;
#pragma unroll
  for (int v = 0; v < 8; ++v) {
    b[2 * v]     = p[(size_t)(2 * v) * ld];
    b[2 * v + 1] = p[(size_t)(2 * v + 1) * ld];
  }
  return b;
}

// Same B layout, converting from f32 row-major source (weight matrices).
__device__ __forceinline__ v16h load_b_f32(const float* __restrict__ src,
                                           int ld, int k0, int n0, int lane) {
  const int col = n0 + (lane & 15);
  const int kb  = (lane < 16) ? 0 : 16;
  const float* p = src + (size_t)(k0 + kb) * ld + col;
  v16h b;
#pragma unroll
  for (int v = 0; v < 8; ++v) {
    b[2 * v]     = (_Float16)p[(size_t)(2 * v) * ld];
    b[2 * v + 1] = (_Float16)p[(size_t)(2 * v + 1) * ld];
  }
  return b;
}

// B-matrix 32x16 where B = M^T, M stored row-major f16 [N x ld] (K along ld).
__device__ __forceinline__ v16h load_bt_f16(const _Float16* __restrict__ src,
                                            int ld, int k0, int n0, int lane) {
  const int col = n0 + (lane & 15);
  const int kb  = (lane < 16) ? 0 : 16;
  const _Float16* p = src + (size_t)col * ld + k0 + kb;
  v16h b;
#pragma unroll
  for (int v = 0; v < 8; ++v) {
    b[2 * v]     = p[2 * v];
    b[2 * v + 1] = p[2 * v + 1];
  }
  return b;
}

#define WMMA_F16(a, b, c) \
  __builtin_amdgcn_wmma_f32_16x16x32_f16(false, (a), false, (b), (short)0, (c), false, false)

// ---------------------------------------------------------------------------
// Kernel 1: projection X[8192x1024] @ W[1024x1024] + bias -> head-layout f16
// ---------------------------------------------------------------------------
__global__ void __launch_bounds__(256) qkv_proj_kernel(
    const float* __restrict__ X, const float* __restrict__ W,
    const float* __restrict__ bias, _Float16* __restrict__ dst) {
  const int lane = threadIdx.x & 31;
  const int wave = threadIdx.x >> 5;
  const int m0   = (blockIdx.x * 8 + wave) * 16;  // row in [0, 8192)
  const int h    = blockIdx.y;                    // head = output-col block of 64
  const int n0   = h * 64;

  v8f acc[4] = {};
  for (int k0 = 0; k0 < D_MODEL; k0 += 32) {
    v16h a = load_a_f32(X, D_MODEL, m0, k0, lane);
#pragma unroll
    for (int j = 0; j < 4; ++j) {
      v16h b = load_b_f32(W, D_MODEL, k0, n0 + 16 * j, lane);
      acc[j] = WMMA_F16(a, b, acc[j]);
    }
  }
#pragma unroll
  for (int j = 0; j < 4; ++j) {
    const int dcol = 16 * j + (lane & 15);
    const float bv = bias[n0 + dcol];
#pragma unroll
    for (int r = 0; r < 8; ++r) {
      const int row = m0 + r + ((lane < 16) ? 0 : 8);
      const int b_  = row >> 11;
      const int s_  = row & 2047;
      dst[(((size_t)b_ * NUM_HEADS + h) * SEQ + s_) * DEPTH + dcol] =
          (_Float16)(acc[j][r] + bv);
    }
  }
}

// ---------------------------------------------------------------------------
// Fused attention kernel: logits -> softmax -> AV, one workgroup per
// (z = b*H + h, 16-query-row block). Logits slab lives in LDS; attn is
// written to HBM exactly once (b128 non-temporal streaming stores).
// ---------------------------------------------------------------------------
__global__ void __launch_bounds__(256) fused_attn_kernel(
    const _Float16* __restrict__ qh, const _Float16* __restrict__ kh,
    const _Float16* __restrict__ vh, const float* __restrict__ mask,
    float* __restrict__ attn, _Float16* __restrict__ ctx) {
  __shared__ float ltile[16 * LSTRIDE];   // 16 x 2048 f32 logits/probs (131 KB)
  __shared__ float redbuf[4 * 16 * 16];   // AV cross-wave partial sums (4 KB)

  const int lane = threadIdx.x & 31;
  const int wave = threadIdx.x >> 5;
  const int z    = blockIdx.y;            // b*H + h
  const int b_   = z >> 4;
  const int h    = z & 15;
  const int m0   = blockIdx.x * 16;       // query-row block within (b,h)

  const _Float16* Q = qh + (size_t)z * SEQ * DEPTH;
  const _Float16* K = kh + (size_t)z * SEQ * DEPTH;
  const _Float16* V = vh + (size_t)z * SEQ * DEPTH;
  float* attnZ      = attn + (size_t)z * SEQ * SEQ;

  // ---- Phase 1: logits tile 16 x 2048; wave w owns 256-col slab ----
  {
    const v16h a0 = load_a_f16(Q, DEPTH, m0, 0, lane);
    const v16h a1 = load_a_f16(Q, DEPTH, m0, 32, lane);
    const int slab = wave * 256;
#pragma unroll 4
    for (int sub = 0; sub < 16; ++sub) {
      const int n0 = slab + sub * 16;
      v8f acc = {};
      acc = WMMA_F16(a0, load_bt_f16(K, DEPTH, 0, n0, lane), acc);
      acc = WMMA_F16(a1, load_bt_f16(K, DEPTH, 32, n0, lane), acc);
      const int col  = n0 + (lane & 15);
      const float mv = mask[b_ * SEQ + col] * -1e9f;
      const int rb   = (lane < 16) ? 0 : 8;
#pragma unroll
      for (int r = 0; r < 8; ++r)
        ltile[(r + rb) * LSTRIDE + col] = acc[r] * 0.125f + mv;
    }
  }
  __syncthreads();

  // ---- Phase 2: softmax; wave w owns rows 2w, 2w+1 ----
  // All sweeps are b128 per lane (ds_load_b128 / ds_store_b128 /
  // global_store_b128 NT) : lane covers 4 consecutive words.
#pragma unroll
  for (int rr = 0; rr < 2; ++rr) {
    const int row = wave * 2 + rr;
    float* p = &ltile[row * LSTRIDE];

    float m = -3.0e38f;
#pragma unroll 4
    for (int it = 0; it < SEQ / 128; ++it) {
      const v4f t = *(const v4f*)&p[it * 128 + lane * 4];
      m = fmaxf(m, fmaxf(fmaxf(t.x, t.y), fmaxf(t.z, t.w)));
    }
#pragma unroll
    for (int off = 16; off > 0; off >>= 1) m = fmaxf(m, __shfl_xor(m, off));

    float sum = 0.0f;
#pragma unroll 4
    for (int it = 0; it < SEQ / 128; ++it) {
      v4f t = *(const v4f*)&p[it * 128 + lane * 4];
      t.x = __expf(t.x - m);
      t.y = __expf(t.y - m);
      t.z = __expf(t.z - m);
      t.w = __expf(t.w - m);
      *(v4f*)&p[it * 128 + lane * 4] = t;
      sum += (t.x + t.y) + (t.z + t.w);
    }
#pragma unroll
    for (int off = 16; off > 0; off >>= 1) sum += __shfl_xor(sum, off);
    const float inv = 1.0f / sum;

    float* gout = attnZ + (size_t)(m0 + row) * SEQ;
#pragma unroll 4
    for (int it = 0; it < SEQ / 128; ++it) {
      v4f t = *(const v4f*)&p[it * 128 + lane * 4];
      t.x *= inv; t.y *= inv; t.z *= inv; t.w *= inv;
      *(v4f*)&p[it * 128 + lane * 4] = t;                        // feeds phase 3
      __builtin_nontemporal_store(t, (v4f*)&gout[it * 128 + lane * 4]);  // HBM, NT
    }
  }
  __syncthreads();

  // ---- Phase 3: ctx(16x64) = probs(16x2048, LDS) @ V(2048x64) ----
  // wave w: n-subtile j = w&3, K-half = w>>2 (1024 each, 32 steps)
  {
    const int j     = wave & 3;
    const int khalf = wave >> 2;
    const int rb    = (lane < 16) ? 0 : 8;
    v8f acc = {};
    for (int k0 = khalf * 1024; k0 < khalf * 1024 + 1024; k0 += 32) {
      // A-frag from LDS f32 probs (cvt to f16)
      const int kb = (lane < 16) ? 0 : 16;
      const float* p = &ltile[(lane & 15) * LSTRIDE + k0 + kb];
      v16h a;
#pragma unroll
      for (int v = 0; v < 8; ++v) {
        const int ko = 2 * v;
        a[2 * v]     = (_Float16)p[ko];
        a[2 * v + 1] = (_Float16)p[ko + 1];
      }
      acc = WMMA_F16(a, load_b_f16(V, DEPTH, k0, 16 * j, lane), acc);
    }
    if (khalf == 1) {
#pragma unroll
      for (int r = 0; r < 8; ++r)
        redbuf[j * 256 + (r + rb) * 16 + (lane & 15)] = acc[r];
    }
    __syncthreads();
    if (khalf == 0) {
#pragma unroll
      for (int r = 0; r < 8; ++r) {
        const int rowl = r + rb;
        const float val = acc[r] + redbuf[j * 256 + rowl * 16 + (lane & 15)];
        const int sq = m0 + rowl;
        const int d  = 16 * j + (lane & 15);
        ctx[(((size_t)b_ * SEQ + sq) * NUM_HEADS + h) * DEPTH + d] = (_Float16)val;
      }
    }
  }
}

// ---------------------------------------------------------------------------
// Kernel 5: out = ctx[8192x1024] @ wo[1024x1024] + bo, f32 row-major output
// ---------------------------------------------------------------------------
__global__ void __launch_bounds__(256) out_proj_kernel(
    const _Float16* __restrict__ ctx, const float* __restrict__ wo,
    const float* __restrict__ bo, float* __restrict__ out) {
  const int lane = threadIdx.x & 31;
  const int wave = threadIdx.x >> 5;
  const int m0   = (blockIdx.x * 8 + wave) * 16;
  const int n0   = blockIdx.y * 64;

  v8f acc[4] = {};
  for (int k0 = 0; k0 < D_MODEL; k0 += 32) {
    v16h a = load_a_f16(ctx, D_MODEL, m0, k0, lane);
#pragma unroll
    for (int j = 0; j < 4; ++j) {
      v16h b = load_b_f32(wo, D_MODEL, k0, n0 + 16 * j, lane);
      acc[j] = WMMA_F16(a, b, acc[j]);
    }
  }
#pragma unroll
  for (int j = 0; j < 4; ++j) {
    const int col  = n0 + 16 * j + (lane & 15);
    const float bv = bo[col];
#pragma unroll
    for (int r = 0; r < 8; ++r) {
      const int row = m0 + r + ((lane < 16) ? 0 : 8);
      // `out` is written once and never re-read by this pipeline -> NT store
      __builtin_nontemporal_store(acc[j][r] + bv,
                                  &out[(size_t)row * D_MODEL + col]);
    }
  }
}

// ---------------------------------------------------------------------------
extern "C" void kernel_launch(void* const* d_in, const int* in_sizes, int n_in,
                              void* d_out, int out_size, void* d_ws, size_t ws_size,
                              hipStream_t stream) {
  const float* v    = (const float*)d_in[0];
  const float* k    = (const float*)d_in[1];
  const float* q    = (const float*)d_in[2];
  const float* mask = (const float*)d_in[3];
  const float* wq   = (const float*)d_in[4];
  const float* bq   = (const float*)d_in[5];
  const float* wk   = (const float*)d_in[6];
  const float* bk   = (const float*)d_in[7];
  const float* wv   = (const float*)d_in[8];
  const float* bv   = (const float*)d_in[9];
  const float* wo   = (const float*)d_in[10];
  const float* bo   = (const float*)d_in[11];

  float* out  = (float*)d_out;                          // [B, S, D]
  float* attn = out + (size_t)BATCH * SEQ * D_MODEL;    // [B, H, S, S]

  const size_t HEAD_ELEMS = (size_t)BATCH * NUM_HEADS * SEQ * DEPTH;
  _Float16* qh  = (_Float16*)d_ws;
  _Float16* kh  = qh + HEAD_ELEMS;
  _Float16* vh  = kh + HEAD_ELEMS;
  _Float16* ctx = vh + HEAD_ELEMS;   // total ws use: 67 MB

  const dim3 blk(256);

  // QKV projections
  const dim3 gProj(ROWS / 128, NUM_HEADS);
  qkv_proj_kernel<<<gProj, blk, 0, stream>>>(q, wq, bq, qh);
  qkv_proj_kernel<<<gProj, blk, 0, stream>>>(k, wk, bk, kh);
  qkv_proj_kernel<<<gProj, blk, 0, stream>>>(v, wv, bv, vh);

  // Fused logits + softmax + AV (attn written to HBM exactly once, NT)
  const dim3 gAttn(SEQ / 16, BATCH * NUM_HEADS);
  fused_attn_kernel<<<gAttn, blk, 0, stream>>>(qh, kh, vh, mask, attn, ctx);

  // Output projection
  const dim3 gOut(ROWS / 128, D_MODEL / 64);
  out_proj_kernel<<<gOut, blk, 0, stream>>>(ctx, wo, bo, out);
}
